// StringModel_52639119179973
// MI455X (gfx1250) — matrix operand order repaired and checked by reading, not compiled, not run on hardware
//
#include <hip/hip_runtime.h>

// ---------------------------------------------------------------- constants
static constexpr int L_  = 4;
static constexpr int B_  = 8;
static constexpr int S_  = 1024;
static constexpr int D_  = 256;
static constexpr int H_  = 8;
static constexpr int DH_ = 32;
static constexpr int I_  = 1024;
static constexpr int M_  = 4;
static constexpr int N_  = B_ * S_;   // 8192 tokens

typedef _Float16 half16v __attribute__((ext_vector_type(16)));
typedef _Float16 half8v  __attribute__((ext_vector_type(8)));
typedef float    float8v __attribute__((ext_vector_type(8)));
typedef unsigned int u32x4 __attribute__((ext_vector_type(4)));
typedef int          i32x4 __attribute__((ext_vector_type(4)));
typedef int          i32x8 __attribute__((ext_vector_type(8)));

#define WMMA_F16 __builtin_amdgcn_wmma_f32_16x16x32_f16

// A fragment (16x32 f16, M x K): lane<16 -> row=lane, K=0..15;
// lane>=16 -> row=lane-16, K=16..31. base points at X[row] + kb.
__device__ __forceinline__ half16v load_a_frag(const _Float16* __restrict__ base, int ksel) {
    half8v lo = *(const half8v*)(base + ksel * 16);
    half8v hi = *(const half8v*)(base + ksel * 16 + 8);
    half16v a;
#pragma unroll
    for (int i = 0; i < 8; ++i) { a[i] = lo[i]; a[8 + i] = hi[i]; }
    return a;
}

// B fragment (32x16 f16, K x N): lanes 0-15 hold K=0..15 of column N=lane,
// lanes 16-31 hold K=16..31 of column N=lane-16; 16 contiguous halves at p.
__device__ __forceinline__ half16v load_b_frag(const _Float16* p) {
    half8v lo = *(const half8v*)p;
    half8v hi = *(const half8v*)(p + 8);
    half16v b;
#pragma unroll
    for (int i = 0; i < 8; ++i) { b[i] = lo[i]; b[8 + i] = hi[i]; }
    return b;
}

// ---------------------------------------------------------------- TDM helper
// 2D tile load global->LDS via Tensor Data Mover (CDNA5), D# per ISA ch.8:
// group0: count=1 | lds_addr | global_addr | type=2
// group1: data_size=1(2B), pad every 64 dwords by 4 dwords (bank de-skew),
//         tensor_dim0/1, tile_dim0/1, tensor_dim0_stride.
__device__ __forceinline__ void tdm_load_2d(unsigned lds_off, const _Float16* g,
                                            unsigned dim0, unsigned dim1,
                                            unsigned til0, unsigned til1,
                                            unsigned long long stride0) {
    unsigned long long ga = (unsigned long long)(uintptr_t)g;
    u32x4 g0;
    g0[0] = 1u;                                        // count=1 (valid user D#)
    g0[1] = lds_off;                                   // lds_addr (bytes)
    g0[2] = (unsigned)ga;                              // global_addr[31:0]
    g0[3] = (unsigned)(ga >> 32) | (2u << 30);         // global_addr[56:32] | type=2
    i32x8 g1;
    g1[0] = (int)((1u << 16) | (1u << 20) | (5u << 22) | (3u << 25));
    //              data_size=2B  pad_en     ival=64dw     amt=4dw
    g1[1] = (int)((dim0 & 0xFFFFu) << 16);             // tensor_dim0[15:0]
    g1[2] = (int)((dim0 >> 16) | ((dim1 & 0xFFFFu) << 16));
    g1[3] = (int)((dim1 >> 16) | (til0 << 16));        // tile_dim0
    g1[4] = (int)(til1 & 0xFFFFu);                     // tile_dim1 (tile_dim2=0)
    g1[5] = (int)(unsigned)stride0;                    // dim0_stride[31:0]
    g1[6] = (int)(unsigned)((stride0 >> 32) & 0xFFFFu);// dim0_stride[47:32]
    g1[7] = 0;
    i32x4 z4 = {0, 0, 0, 0};
#if __clang_major__ >= 23
    i32x8 z8 = {0, 0, 0, 0, 0, 0, 0, 0};
    __builtin_amdgcn_tensor_load_to_lds(g0, g1, z4, z4, z8, 0);
#else
    __builtin_amdgcn_tensor_load_to_lds(g0, g1, z4, z4, 0);
#endif
}

// ---------------------------------------------------------------- elementwise
__global__ void cvt_f32_to_f16(const float* __restrict__ in, _Float16* __restrict__ out, int n) {
    int i = blockIdx.x * blockDim.x + threadIdx.x;
    if (i < n) out[i] = (_Float16)in[i];
}

__global__ void embed_kernel(const int* __restrict__ x, const float* __restrict__ emb_w,
                             const float* __restrict__ pe,
                             float* __restrict__ h32, _Float16* __restrict__ h16) {
    int idx = blockIdx.x * blockDim.x + threadIdx.x;      // [N_*D_]
    int tok = idx >> 8;            // /256
    int d   = idx & 255;
    int s   = tok & (S_ - 1);      // tok % S
    float v = emb_w[x[tok] * D_ + d] + pe[s * D_ + d];
    h32[idx] = v;
    h16[idx] = (_Float16)v;
}

// v16 [B,S,H,DH] -> vt16 [B,H,DH,S]
__global__ void transpose_v(const _Float16* __restrict__ v, _Float16* __restrict__ vt) {
    int idx = blockIdx.x * blockDim.x + threadIdx.x;      // [B*S*H*DH] = 2^21
    int dh = idx & 31;
    int h  = (idx >> 5) & 7;
    int s  = (idx >> 8) & 1023;
    int b  = idx >> 18;
    vt[(((size_t)b * H_ + h) * DH_ + dh) * S_ + s] = v[idx];
}

// ---------------------------------------------------------------- WMMA GEMM (TDM-staged B)
// Y[n][j] = sum_k X[n][k] * W[j][k] + bias[j]   (torch Linear: X @ W.T + b)
// Block = 8 waves: 256(M) x 64(N) output. The 64-row weight strip is shared by
// all waves, so wave 0 streams it through LDS with TENSOR_LOAD_TO_LDS in
// double-buffered 64x128 f16 chunks (TENSORcnt pipelined); each wave computes
// a 32x64 block (2 A-frags from global x 4 B-frags from LDS -> 8 WMMAs).
static constexpr int CH_   = 128;            // K-chunk (f16 elements)
static constexpr int BPAD_ = CH_ + 8;        // LDS row stride incl. TDM padding
__global__ void gemm16t(const _Float16* __restrict__ X, const _Float16* __restrict__ W,
                        const float* __restrict__ bias,
                        float* __restrict__ out32, _Float16* __restrict__ out16,
                        int Nrows, int K, int Dout, int act_gelu) {
    __shared__ _Float16 bsm[2][64 * BPAD_];  // 2 x 17KB double buffer

    const int lane = threadIdx.x & 31;
    const int wv   = threadIdx.x >> 5;
    const int nb_n = Dout >> 6;              // # of 64-col strips
    const int nb   = blockIdx.x % nb_n;      // strip: shared by whole block
    const int mb   = (blockIdx.x / nb_n) * 8 + wv;  // 32-row block per wave

    const int l16  = lane & 15;
    const int ksel = lane >> 4;
    const size_t arow0 = (size_t)(mb * 32 + l16) * K;
    const size_t arow1 = arow0 + (size_t)16 * K;
    const _Float16* wstrip = W + (size_t)(nb * 64) * K;

    float8v acc[2][4];
#pragma unroll
    for (int i = 0; i < 2; ++i)
#pragma unroll
        for (int j = 0; j < 4; ++j)
            acc[i][j] = (float8v){0.f,0.f,0.f,0.f,0.f,0.f,0.f,0.f};

    const int nchunks = K / CH_;
    if (wv == 0)
        tdm_load_2d((unsigned)(uintptr_t)&bsm[0][0], wstrip,
                    (unsigned)K, 64u, CH_, 64u, (unsigned long long)K);

    for (int c = 0; c < nchunks; ++c) {
        if (wv == 0) {
            if (c + 1 < nchunks) {
                tdm_load_2d((unsigned)(uintptr_t)&bsm[(c + 1) & 1][0],
                            wstrip + (size_t)(c + 1) * CH_,
                            (unsigned)K, 64u, CH_, 64u, (unsigned long long)K);
                // TDM ops complete in order: cnt<=1 means chunk c has landed.
                __builtin_amdgcn_s_wait_tensorcnt(1);
            } else {
                __builtin_amdgcn_s_wait_tensorcnt(0);
            }
        }
        __syncthreads();

        const _Float16* bt = &bsm[c & 1][0];
#pragma unroll
        for (int kb2 = 0; kb2 < CH_; kb2 += 32) {
            const int kb = c * CH_ + kb2;
            __builtin_prefetch(X + arow0 + kb + 32, 0, 1);  // global_prefetch_b8
            half16v a0 = load_a_frag(X + arow0 + kb, ksel);
            half16v a1 = load_a_frag(X + arow1 + kb, ksel);
            half16v bf[4];
#pragma unroll
            for (int j = 0; j < 4; ++j)
                bf[j] = load_b_frag(bt + (j * 16 + l16) * BPAD_ + kb2 + ksel * 16);
#pragma unroll
            for (int j = 0; j < 4; ++j) {
                acc[0][j] = WMMA_F16(false, a0, false, bf[j], (short)0, acc[0][j], false, false);
                acc[1][j] = WMMA_F16(false, a1, false, bf[j], (short)0, acc[1][j], false, false);
            }
        }
        __syncthreads();   // buffer (c&1) free for reuse before next TDM issue
    }

#pragma unroll
    for (int i = 0; i < 2; ++i) {
#pragma unroll
        for (int j = 0; j < 4; ++j) {
            const int col = nb * 64 + j * 16 + l16;
            const float bv = bias ? bias[col] : 0.f;
#pragma unroll
            for (int r = 0; r < 8; ++r) {
                int row = mb * 32 + i * 16 + r + (ksel << 3);  // C/D layout
                float v = acc[i][j][r] + bv;
                if (act_gelu) v = 0.5f * v * (1.0f + erff(v * 0.70710678118f));
                size_t o = (size_t)row * Dout + col;
                if (out32) out32[o] = v;
                if (out16) out16[o] = (_Float16)v;
            }
        }
    }
}

// ---------------------------------------------------------------- flash attention
// One wave per (b, h, 16-query tile). Q,K: [B,S,H,DH] f16. Vt: [B,H,DH,S] f16.
// ctx out: [B,S,H,DH] f16 (== [B,S,D]).
static constexpr int AW_ = 4;   // waves per block
__global__ void attn_flash(const _Float16* __restrict__ Q, const _Float16* __restrict__ Kt,
                           const _Float16* __restrict__ Vt, _Float16* __restrict__ ctx) {
    __shared__ _Float16 lds_p[AW_][16][36];   // per-wave 16x32 prob tile (+pad)

    const int lane = threadIdx.x & 31;
    const int wv   = threadIdx.x >> 5;
    const int gw   = blockIdx.x * AW_ + wv;
    const int qt   = gw % (S_ / 16);
    const int bh   = gw / (S_ / 16);
    const int b    = bh / H_;
    const int h    = bh % H_;
    const int qb   = qt << 4;
    const int l16  = lane & 15;
    const int ksel = lane >> 4;

    // Q A-fragment (rows = 16 queries, K = DH=32)
    const _Float16* qp = Q + (((size_t)b * S_ + (qb + l16)) * H_ + h) * DH_;
    half16v qa = load_a_frag(qp, ksel);

    float8v acc0 = {0.f,0.f,0.f,0.f,0.f,0.f,0.f,0.f};
    float8v acc1 = {0.f,0.f,0.f,0.f,0.f,0.f,0.f,0.f};
    float mrow[8], lrow[8];
#pragma unroll
    for (int r = 0; r < 8; ++r) { mrow[r] = -1e30f; lrow[r] = 0.f; }

    const float scale = 0.17677669529663687f;   // 1/sqrt(32)
    const int nkeys = qb + 16;                  // causal: keys 0 .. qb+15

    for (int kb = 0; kb < nkeys; kb += 32) {
        // --- scores: two 16x16 tiles via WMMA (B = K rows, dh contiguous)
        float8v sc[2];
#pragma unroll
        for (int t = 0; t < 2; ++t) {
            int key = kb + t * 16 + l16;
            const _Float16* kp = Kt + (((size_t)b * S_ + key) * H_ + h) * DH_ + ksel * 16;
            half16v bf = load_b_frag(kp);
            float8v z = {0.f,0.f,0.f,0.f,0.f,0.f,0.f,0.f};
            sc[t] = WMMA_F16(false, qa, false, bf, (short)0, z, false, false);
        }
        // --- online softmax in fp32 (C layout: col=l16, row = r + 8*ksel)
#pragma unroll
        for (int r = 0; r < 8; ++r) {
            int grow = qb + r + (ksel << 3);
            float s0 = sc[0][r] * scale + (((kb + l16)      > grow) ? -1e9f : 0.f);
            float s1 = sc[1][r] * scale + (((kb + 16 + l16) > grow) ? -1e9f : 0.f);
            float mx = fmaxf(s0, s1);
#pragma unroll
            for (int off = 1; off < 16; off <<= 1) mx = fmaxf(mx, __shfl_xor(mx, off, 32));
            float nm = fmaxf(mrow[r], mx);
            float p0 = __expf(s0 - nm);
            float p1 = __expf(s1 - nm);
            float corr = __expf(mrow[r] - nm);
            float ps = p0 + p1;
#pragma unroll
            for (int off = 1; off < 16; off <<= 1) ps += __shfl_xor(ps, off, 32);
            lrow[r] = lrow[r] * corr + ps;
            mrow[r] = nm;
            acc0[r] *= corr;
            acc1[r] *= corr;
            lds_p[wv][r + (ksel << 3)][l16]      = (_Float16)p0;
            lds_p[wv][r + (ksel << 3)][16 + l16] = (_Float16)p1;
        }
        // --- re-layout probs (C layout -> A fragment) through per-wave LDS
        half16v pa;
#pragma unroll
        for (int j = 0; j < 8; ++j) {
            pa[j]     = lds_p[wv][l16][(ksel << 4) + j];
            pa[8 + j] = lds_p[wv][l16][(ksel << 4) + 8 + j];
        }
        // --- ctx += P(16x32keys) x V(32keys x 32dh), two N-tiles of 16
#pragma unroll
        for (int t = 0; t < 2; ++t) {
            int dh = t * 16 + l16;
            const _Float16* vp = Vt + (((size_t)b * H_ + h) * DH_ + dh) * S_ + kb + ksel * 16;
            half16v bf = load_b_frag(vp);
            if (t == 0) acc0 = WMMA_F16(false, pa, false, bf, (short)0, acc0, false, false);
            else        acc1 = WMMA_F16(false, pa, false, bf, (short)0, acc1, false, false);
        }
    }

#pragma unroll
    for (int r = 0; r < 8; ++r) {
        int grow = qb + r + (ksel << 3);
        float inv = 1.f / lrow[r];
        size_t base = (((size_t)b * S_ + grow) * H_ + h) * DH_;
        ctx[base + l16]      = (_Float16)(acc0[r] * inv);
        ctx[base + 16 + l16] = (_Float16)(acc1[r] * inv);
    }
}

// ---------------------------------------------------------------- layernorm (+residual)
__global__ void ln_kernel(const float* __restrict__ pre, const float* __restrict__ resid,
                          const float* __restrict__ g, const float* __restrict__ bta,
                          float* __restrict__ out32, _Float16* __restrict__ out16) {
    __shared__ float red[D_];
    const int row = blockIdx.x;
    const int t   = threadIdx.x;
    const size_t o = (size_t)row * D_ + t;
    float v = pre[o] + resid[o];
    red[t] = v;
    __syncthreads();
    for (int s = D_ / 2; s > 0; s >>= 1) { if (t < s) red[t] += red[t + s]; __syncthreads(); }
    float mean = red[0] * (1.f / D_);
    __syncthreads();
    float d = v - mean;
    red[t] = d * d;
    __syncthreads();
    for (int s = D_ / 2; s > 0; s >>= 1) { if (t < s) red[t] += red[t + s]; __syncthreads(); }
    float var = red[0] * (1.f / D_);
    float y = g[t] * d * rsqrtf(var + 1e-12f) + bta[t];
    out32[o] = y;
    out16[o] = (_Float16)y;
}

// ---------------------------------------------------------------- logits (M=4, scalar fp32)
__global__ void logits_kernel(const float* __restrict__ h32, const float* __restrict__ out_w,
                              const float* __restrict__ out_b, float* __restrict__ out) {
    int idx = blockIdx.x * blockDim.x + threadIdx.x;   // [N_*M_]
    int tok = idx >> 2;
    int m   = idx & 3;
    const float* hp = h32 + (size_t)tok * D_;
    const float* wp = out_w + (size_t)m * D_;
    float s = out_b[m];
#pragma unroll 4
    for (int d = 0; d < D_; ++d) s += hp[d] * wp[d];
    out[idx] = s;
}

// ---------------------------------------------------------------- launch
extern "C" void kernel_launch(void* const* d_in, const int* in_sizes, int n_in,
                              void* d_out, int out_size, void* d_ws, size_t ws_size,
                              hipStream_t stream) {
    (void)in_sizes; (void)n_in; (void)out_size; (void)ws_size;

    const int*   x     = (const int*)  d_in[0];
    const float* emb_w = (const float*)d_in[1];
    const float* pe    = (const float*)d_in[2];
    const float* q_w   = (const float*)d_in[3];
    const float* q_b   = (const float*)d_in[4];
    const float* k_w   = (const float*)d_in[5];
    const float* k_b   = (const float*)d_in[6];
    const float* v_w   = (const float*)d_in[7];
    const float* v_b   = (const float*)d_in[8];
    const float* o_w   = (const float*)d_in[9];
    const float* o_b   = (const float*)d_in[10];
    const float* ln1_g = (const float*)d_in[11];
    const float* ln1_b = (const float*)d_in[12];
    const float* f1_w  = (const float*)d_in[13];
    const float* f1_b  = (const float*)d_in[14];
    const float* f2_w  = (const float*)d_in[15];
    const float* f2_b  = (const float*)d_in[16];
    const float* ln2_g = (const float*)d_in[17];
    const float* ln2_b = (const float*)d_in[18];
    const float* out_w = (const float*)d_in[19];
    const float* out_b = (const float*)d_in[20];
    float* logits = (float*)d_out;

    // ---- workspace carve (256B aligned)
    size_t off = 0;
    auto carve = [&](size_t bytes) -> void* {
        void* p = (char*)d_ws + off;
        off += (bytes + 255) & ~(size_t)255;
        return p;
    };
    float*    h32   = (float*)   carve((size_t)N_ * D_ * 4);
    _Float16* h16   = (_Float16*)carve((size_t)N_ * D_ * 2);
    float*    a32   = (float*)   carve((size_t)N_ * D_ * 4);
    _Float16* a16   = (_Float16*)carve((size_t)N_ * D_ * 2);
    _Float16* q16   = (_Float16*)carve((size_t)N_ * D_ * 2);
    _Float16* k16   = (_Float16*)carve((size_t)N_ * D_ * 2);
    _Float16* v16   = (_Float16*)carve((size_t)N_ * D_ * 2);
    _Float16* vt16  = (_Float16*)carve((size_t)N_ * D_ * 2);
    _Float16* ctx16 = (_Float16*)carve((size_t)N_ * D_ * 2);
    float*    pre32 = (float*)   carve((size_t)N_ * D_ * 4);
    _Float16* ff16  = (_Float16*)carve((size_t)N_ * I_ * 2);
    _Float16* qw16  = (_Float16*)carve((size_t)L_ * D_ * D_ * 2);
    _Float16* kw16  = (_Float16*)carve((size_t)L_ * D_ * D_ * 2);
    _Float16* vw16  = (_Float16*)carve((size_t)L_ * D_ * D_ * 2);
    _Float16* ow16  = (_Float16*)carve((size_t)L_ * D_ * D_ * 2);
    _Float16* f1w16 = (_Float16*)carve((size_t)L_ * I_ * D_ * 2);
    _Float16* f2w16 = (_Float16*)carve((size_t)L_ * D_ * I_ * 2);

    // ---- weight conversion to f16 (cheap; rerun each call for determinism)
    auto cvt = [&](const float* src, _Float16* dst, int n) {
        cvt_f32_to_f16<<<(n + 255) / 256, 256, 0, stream>>>(src, dst, n);
    };
    cvt(q_w, qw16, L_ * D_ * D_);
    cvt(k_w, kw16, L_ * D_ * D_);
    cvt(v_w, vw16, L_ * D_ * D_);
    cvt(o_w, ow16, L_ * D_ * D_);
    cvt(f1_w, f1w16, L_ * I_ * D_);
    cvt(f2_w, f2w16, L_ * D_ * I_);

    // ---- embedding + positional encoding
    embed_kernel<<<(N_ * D_) / 256, 256, 0, stream>>>(x, emb_w, pe, h32, h16);

    auto gemm = [&](const _Float16* X, const _Float16* W, const float* bias,
                    float* o32, _Float16* o16, int Nr, int K, int Do, int gelu) {
        int blocks = (Nr / 256) * (Do / 64);   // block: 8 waves, 256x64 output
        gemm16t<<<blocks, 256, 0, stream>>>(X, W, bias, o32, o16, Nr, K, Do, gelu);
    };

    for (int l = 0; l < L_; ++l) {
        const size_t wo = (size_t)l * D_ * D_;
        // QKV projections (f16 WMMA, fp32 accum, TDM-staged weights)
        gemm(h16, qw16 + wo, q_b + l * D_, nullptr, q16, N_, D_, D_, 0);
        gemm(h16, kw16 + wo, k_b + l * D_, nullptr, k16, N_, D_, D_, 0);
        gemm(h16, vw16 + wo, v_b + l * D_, nullptr, v16, N_, D_, D_, 0);
        // V -> [B,H,DH,S] for contiguous ctx B-fragments
        transpose_v<<<(N_ * D_) / 256, 256, 0, stream>>>(v16, vt16);
        // flash attention: wave per (b,h,qtile)
        attn_flash<<<(B_ * H_ * (S_ / 16)) / AW_, AW_ * 32, 0, stream>>>(q16, k16, vt16, ctx16);
        // output projection -> pre32; LN(pre + h) -> a
        gemm(ctx16, ow16 + wo, o_b + l * D_, pre32, nullptr, N_, D_, D_, 0);
        ln_kernel<<<N_, D_, 0, stream>>>(pre32, h32, ln1_g + l * D_, ln1_b + l * D_, a32, a16);
        // FFN: gelu(a @ f1.T + b1) @ f2.T + b2; LN(pre + a) -> h
        gemm(a16, f1w16 + (size_t)l * I_ * D_, f1_b + l * I_, nullptr, ff16, N_, D_, I_, 1);
        gemm(ff16, f2w16 + (size_t)l * D_ * I_, f2_b + l * D_, pre32, nullptr, N_, I_, D_, 0);
        ln_kernel<<<N_, D_, 0, stream>>>(pre32, a32, ln2_g + l * D_, ln2_b + l * D_, h32, h16);
    }

    // final logits in fp32 (Dout=4 < WMMA tile; scalar is appropriate)
    logits_kernel<<<(N_ * M_) / 256, 256, 0, stream>>>(h32, out_w, out_b, logits);
}
